// TransformerBackbone_13941463842852
// MI455X (gfx1250) — matrix operand order repaired and checked by reading, not compiled, not run on hardware
//
#include <hip/hip_runtime.h>
#include <hip/hip_bf16.h>
#include <cstdint>
#include <cstddef>

typedef _Float16 f16;
typedef __attribute__((ext_vector_type(16))) _Float16 v16h;
typedef __attribute__((ext_vector_type(8)))  _Float16 v8h;
typedef __attribute__((ext_vector_type(8)))  float    v8f;

#define DD   1024
#define HH   16
#define KHH  4
#define HDD  64
#define FFD  3072
#define LL   8
#define BB   2
#define SS   2048
#define WW   128
#define KK   5
#define MTOK (BB*SS)   /* 4096 tokens */

// ---------------------------------------------------------------------------
// CDNA5 async copy global->LDS (ASYNCcnt) helpers
// ---------------------------------------------------------------------------
__device__ inline uint32_t lds32(const void* p) {
  // generic pointer to LDS: aperture lives in the high 32 bits, low 32 = LDS addr
  return (uint32_t)(uintptr_t)p;
}
__device__ inline void async_b128(uint32_t lds_off, const void* g) {
  uint64_t ga = (uint64_t)(uintptr_t)g;
  asm volatile("global_load_async_to_lds_b128 %0, %1, off"
               :: "v"(lds_off), "v"(ga) : "memory");
}
__device__ inline void wait_async0() {
  asm volatile("s_wait_asynccnt 0" ::: "memory");
}

// ---------------------------------------------------------------------------
// WMMA helpers (gfx1250, wave32). A/B frags use contiguous k order, k = 16*hi + s.
// ---------------------------------------------------------------------------
__device__ inline v8f vzero8() {
  v8f z = {0.f,0.f,0.f,0.f,0.f,0.f,0.f,0.f};
  return z;
}
__device__ inline v8f wmma16(v16h a, v16h b, v8f c) {
  return __builtin_amdgcn_wmma_f32_16x16x32_f16(false, a, false, b, (short)0, c, false, false);
}

// A fragment (16x32 f16, MxK) from row-major LDS tile: two contiguous 8-f16 runs
// per lane -> 2x ds_load_b128.  row = lane&15 ; runs at [hi*16 .. +7] and [hi*16+8 .. +7]
__device__ inline v16h frag_a(const f16* base, int ld, int row, int lane) {
  v16h a;
  const int hi = (lane >> 4) & 1;
  const f16* rp = base + (size_t)row * ld + hi * 16;
#pragma unroll
  for (int v0 = 0; v0 < 4; ++v0) { a[2*v0]   = rp[2*v0];    a[2*v0+1] = rp[2*v0+1]; }
#pragma unroll
  for (int v0 = 0; v0 < 4; ++v0) { a[8+2*v0] = rp[8+2*v0]; a[9+2*v0] = rp[9+2*v0]; }
  return a;
}

// B fragment (32x16 f16, KxN) when LDS stores B^T rows (n-major, K contiguous):
// per lane 16 contiguous f16 -> 2x ds_load_b128.
// b[2v] = B[hi*16+2v][n] = Bt[n][hi*16+2v]
__device__ inline v16h frag_bt(const f16* base, int ld, int nrow, int colbase, int lane) {
  v16h b;
  const int hi = (lane >> 4) & 1;
  const f16* rp = base + (size_t)(nrow + (lane & 15)) * ld + colbase + hi * 16;
#pragma unroll
  for (int i = 0; i < 16; ++i) b[i] = rp[i];
  return b;
}

// ---------------------------------------------------------------------------
// RoPE tables: cos/sin [S][32]
// ---------------------------------------------------------------------------
__global__ void k_rope_tables(float* __restrict__ cosT, float* __restrict__ sinT) {
  int idx = blockIdx.x * blockDim.x + threadIdx.x;
  if (idx >= SS * 32) return;
  int j = idx & 31;
  int pos = idx >> 5;
  float inv = __expf(-((float)j / 32.f) * 9.210340371976184f); // ln(10000)
  float f = (float)pos * inv;
  cosT[idx] = cosf(f);
  sinT[idx] = sinf(f);
}

// ---------------------------------------------------------------------------
// RMSNorm: one block per row of 1024; optional f32 and f16 outputs.
// ---------------------------------------------------------------------------
__global__ __launch_bounds__(256) void k_rmsnorm(const float* __restrict__ x,
                                                 const float* __restrict__ w,
                                                 float* yf, f16* yh) {
  const int row = blockIdx.x;
  const int t = threadIdx.x;
  const float* xr = x + (size_t)row * DD;
  float ss = 0.f;
  for (int i = t; i < DD; i += 256) { float v = xr[i]; ss += v * v; }
  __shared__ float red[256];
  red[t] = ss; __syncthreads();
  for (int o = 128; o > 0; o >>= 1) { if (t < o) red[t] += red[t + o]; __syncthreads(); }
  const float nrm = rsqrtf(red[0] * (1.f / (float)DD) + 1e-6f);
  for (int i = t; i < DD; i += 256) {
    float v = xr[i] * nrm * w[i];
    if (yf) yf[(size_t)row * DD + i] = v;
    if (yh) yh[(size_t)row * DD + i] = (f16)v;
  }
}

// ---------------------------------------------------------------------------
// Depthwise causal conv1d over seq, kernel K=5, weights dw[c][t].
// ---------------------------------------------------------------------------
__global__ void k_dwconv(const float* __restrict__ nx, const float* __restrict__ dw,
                         float* __restrict__ out) {
  int idx = blockIdx.x * blockDim.x + threadIdx.x;
  if (idx >= MTOK * DD) return;
  int c = idx % DD;
  int s = (idx / DD) % SS;
  int b = idx / (DD * SS);
  float acc = 0.f;
#pragma unroll
  for (int t = 0; t < KK; ++t) {
    int sp = s + t - (KK - 1);
    if (sp >= 0) acc += nx[((size_t)(b * SS + sp)) * DD + c] * dw[c * KK + t];
  }
  out[idx] = acc;
}

// ---------------------------------------------------------------------------
// f32 -> f16 converts. k_cvt_t writes B^T layout: d[n*K + k] = s[k*N + n]
// ---------------------------------------------------------------------------
__global__ void k_cvt(const float* __restrict__ s, f16* __restrict__ d, int n) {
  int i = blockIdx.x * blockDim.x + threadIdx.x;
  if (i < n) d[i] = (f16)s[i];
}
__global__ void k_cvt_t(const float* __restrict__ s, f16* __restrict__ d, int R, int C) {
  int i = blockIdx.x * blockDim.x + threadIdx.x;
  if (i >= R * C) return;
  int r = i / C, c = i % C;
  d[(size_t)c * R + r] = (f16)s[(size_t)r * C + c];
}

// ---------------------------------------------------------------------------
// RoPE: read f32 projection, write rotated f16 (ready for attention staging)
// ---------------------------------------------------------------------------
__global__ void k_rope(const float* __restrict__ src, f16* __restrict__ dst,
                       const float* __restrict__ cosT, const float* __restrict__ sinT,
                       int nheads) {
  int idx = blockIdx.x * blockDim.x + threadIdx.x;
  if (idx >= MTOK * nheads * 32) return;
  int j = idx & 31;
  int h = (idx >> 5) % nheads;
  int row = idx / (32 * nheads);
  int pos = row % SS;
  float c = cosT[pos * 32 + j], s = sinT[pos * 32 + j];
  size_t base = (size_t)row * (nheads * HDD) + h * HDD;
  float x1 = src[base + j], x2 = src[base + j + 32];
  dst[base + j]      = (f16)(x1 * c + x2 * s);
  dst[base + j + 32] = (f16)(x2 * c - x1 * s);
}

// ---------------------------------------------------------------------------
// WMMA GEMM: C[M,N] = epilogue( A_f16[M,K] @ Bt_f16[N,K]^T )
// 256 threads / 8 waves; block tile 128x64; wave tile 32x32; BK=32.
// Async global->LDS staging, double buffered.
// ---------------------------------------------------------------------------
__global__ __launch_bounds__(256) void k_gemm(const f16* __restrict__ A,
                                              const f16* __restrict__ Bt,
                                              int M, int N, int K,
                                              float* outf, f16* outh,
                                              const float* resid,
                                              const float* scale_ptr,
                                              const float* upre) {
  __shared__ __align__(16) f16 As[2][128][40];   // 80B rows
  __shared__ __align__(16) f16 Bs[2][64][40];    // B^T tile: [n][k]
  const int t    = threadIdx.x;
  const int lane = t & 31;
  const int w    = t >> 5;
  const int wm   = w >> 1, wn = w & 1;
  const int bm   = blockIdx.y * 128;
  const int bn   = blockIdx.x * 64;

  v8f acc[2][2];
  acc[0][0] = vzero8(); acc[0][1] = vzero8();
  acc[1][0] = vzero8(); acc[1][1] = vzero8();

  const int ar = t >> 2, ac = (t & 3) * 8;   // A: 64 rows/pass x 8 f16
  const int nr = t >> 2, nc = (t & 3) * 8;   // Bt: 64 rows x 8 f16

  // prologue: stage tile 0 into buffer 0
  {
    const f16* ag = A + (size_t)(bm + ar) * K + ac;
    async_b128(lds32(&As[0][ar][ac]),      ag);
    async_b128(lds32(&As[0][ar + 64][ac]), ag + (size_t)64 * K);
    async_b128(lds32(&Bs[0][nr][nc]),      Bt + (size_t)(bn + nr) * K + nc);
  }

  int buf = 0;
  for (int k0 = 0; k0 < K; k0 += 32) {
    wait_async0();
    __syncthreads();
    if (k0 + 32 < K) {
      const int nk = k0 + 32, nb = buf ^ 1;
      const f16* ag = A + (size_t)(bm + ar) * K + nk + ac;
      async_b128(lds32(&As[nb][ar][ac]),      ag);
      async_b128(lds32(&As[nb][ar + 64][ac]), ag + (size_t)64 * K);
      async_b128(lds32(&Bs[nb][nr][nc]),      Bt + (size_t)(bn + nr) * K + nk + nc);
    }
    v16h a0 = frag_a(&As[buf][0][0], 40, wm * 32 +      (lane & 15), lane);
    v16h a1 = frag_a(&As[buf][0][0], 40, wm * 32 + 16 + (lane & 15), lane);
    v16h b0 = frag_bt(&Bs[buf][0][0], 40, wn * 32,      0, lane);
    v16h b1 = frag_bt(&Bs[buf][0][0], 40, wn * 32 + 16, 0, lane);

    acc[0][0] = wmma16(a0, b0, acc[0][0]);
    acc[0][1] = wmma16(a0, b1, acc[0][1]);
    acc[1][0] = wmma16(a1, b0, acc[1][0]);
    acc[1][1] = wmma16(a1, b1, acc[1][1]);
    buf ^= 1;
  }

  const float alpha = scale_ptr ? *scale_ptr : 1.0f;
  const int hi = (lane >> 4) & 1, nn = lane & 15;
#pragma unroll
  for (int mi = 0; mi < 2; ++mi)
#pragma unroll
    for (int nj = 0; nj < 2; ++nj)
#pragma unroll
      for (int r = 0; r < 8; ++r) {
        int gm = bm + wm * 32 + mi * 16 + r + 8 * hi;
        int gn = bn + wn * 32 + nj * 16 + nn;
        size_t gi = (size_t)gm * N + gn;
        float val = acc[mi][nj][r] * alpha;
        if (resid) val += resid[gi];
        if (upre)  { float u = upre[gi]; val *= u / (1.0f + __expf(-u)); }
        if (outf) outf[gi] = val;
        if (outh) outh[gi] = (f16)val;
      }
}

// ---------------------------------------------------------------------------
// Windowed attention. One block = (batch, seq-block, head, 32-query tile).
// 128 threads / 4 waves. q/k/v arrive as f16 (rope already applied to q,k).
// Q,K staged via async global->LDS; V staged transposed for contiguous frags.
// ---------------------------------------------------------------------------
__global__ __launch_bounds__(128) void k_attn(const f16* __restrict__ qh,
                                              const f16* __restrict__ khb,
                                              const f16* __restrict__ vhb,
                                              f16* __restrict__ oh) {
  __shared__ __align__(16) f16  Qs[32][72];
  __shared__ __align__(16) f16  Ks[32][72];
  __shared__ __align__(16) f16  Vt[64][40];   // V^T chunk: [d][key]
  __shared__ float              Ss[32][260];
  __shared__ __align__(16) f16  Ps[32][256];

  int id = blockIdx.x;
  const int qt   = id & 3;  id >>= 2;
  const int head = id & 15; id >>= 4;
  const int blk  = id & 15; id >>= 4;
  const int b    = id;
  const int kh   = head >> 2;       // GQA: G=4
  const int t    = threadIdx.x;
  const int lane = t & 31;
  const int w    = t >> 5;
  const int mi   = w >> 1;

  // stage Q (32 x 64) async: thread -> one row quarter (2 x b128)
  {
    int r = t >> 2, c0 = (t & 3) * 16;
    int pos = blk * WW + qt * 32 + r;
    const f16* src = qh + (size_t)(b * SS + pos) * (HH * HDD) + head * HDD + c0;
    async_b128(lds32(&Qs[r][c0]),     src);
    async_b128(lds32(&Qs[r][c0 + 8]), src + 8);
  }

  // ---- scores: S[32 x 256] = Q @ K^T / 8 ----
  for (int kc = 0; kc < 8; ++kc) {
    __syncthreads();
    const bool zero_chunk = (blk == 0) && (kc < 4);
    int i = t >> 2, c0 = (t & 3) * 16;
    if (zero_chunk) {
#pragma unroll
      for (int c = 0; c < 16; ++c) Ks[i][c0 + c] = (f16)0.f;
    } else {
      int kpos = blk * WW + kc * 32 + i - WW;
      const f16* src = khb + (size_t)(b * SS + kpos) * (KHH * HDD) + kh * HDD + c0;
      async_b128(lds32(&Ks[i][c0]),     src);
      async_b128(lds32(&Ks[i][c0 + 8]), src + 8);
    }
    wait_async0();
    __syncthreads();

    const int ni = w & 1;
    v8f sacc = vzero8();
#pragma unroll
    for (int k0 = 0; k0 < HDD; k0 += 32) {
      v16h af = frag_a(&Qs[0][0] + k0, 72, mi * 16 + (lane & 15), lane);
      v16h bf = frag_bt(&Ks[0][0], 72, ni * 16, k0, lane);
      sacc = wmma16(af, bf, sacc);
    }
    const int hi = (lane >> 4) & 1, nn = lane & 15;
#pragma unroll
    for (int r = 0; r < 8; ++r)
      Ss[mi * 16 + r + 8 * hi][kc * 32 + ni * 16 + nn] = sacc[r] * 0.125f;
  }
  __syncthreads();

  // ---- masked softmax per query row ----
  if (t < 32) {
    const int p = qt * 32 + t;  // query position within block [0,128)
    float mx = -1e30f;
    for (int m = 0; m < 256; ++m) {
      bool ok = (m > p) && (m <= p + WW) && (blk > 0 || m >= WW);
      if (ok) mx = fmaxf(mx, Ss[t][m]);
    }
    float sum = 0.f;
    for (int m = 0; m < 256; ++m) {
      bool ok = (m > p) && (m <= p + WW) && (blk > 0 || m >= WW);
      float e = ok ? __expf(Ss[t][m] - mx) : 0.f;
      Ss[t][m] = e;
      sum += e;
    }
    float inv = 1.f / sum;
    for (int m = 0; m < 256; ++m) Ps[t][m] = (f16)(Ss[t][m] * inv);
  }

  // ---- AV: O[32 x 64] = P @ V ----
  v8f oacc0 = vzero8(), oacc1 = vzero8();
  const int nb0 = (w & 1) * 2;
  for (int vc = 0; vc < 8; ++vc) {
    __syncthreads();
    {
      int i = t >> 2, c0 = (t & 3) * 16;     // key i, d range [c0, c0+16)
      const bool zero_chunk = (blk == 0) && (vc < 4);
      if (zero_chunk) {
#pragma unroll
        for (int c = 0; c < 16; ++c) Vt[c0 + c][i] = (f16)0.f;
      } else {
        int kpos = blk * WW + vc * 32 + i - WW;
        const f16* src = vhb + (size_t)(b * SS + kpos) * (KHH * HDD) + kh * HDD + c0;
#pragma unroll
        for (int c = 0; c < 16; ++c) Vt[c0 + c][i] = src[c];
      }
    }
    __syncthreads();
    v16h af = frag_a(&Ps[0][0] + vc * 32, 256, mi * 16 + (lane & 15), lane);
    v16h b0 = frag_bt(&Vt[0][0], 40, (nb0)     * 16, 0, lane);
    v16h b1 = frag_bt(&Vt[0][0], 40, (nb0 + 1) * 16, 0, lane);
    oacc0 = wmma16(af, b0, oacc0);
    oacc1 = wmma16(af, b1, oacc1);
  }

  const int hi = (lane >> 4) & 1, nn = lane & 15;
#pragma unroll
  for (int r = 0; r < 8; ++r) {
    int pos = blk * WW + qt * 32 + mi * 16 + r + 8 * hi;
    size_t gi = (size_t)(b * SS + pos) * (HH * HDD) + head * HDD;
    oh[gi + (nb0)     * 16 + nn] = (f16)oacc0[r];
    oh[gi + (nb0 + 1) * 16 + nn] = (f16)oacc1[r];
  }
}

// ---------------------------------------------------------------------------
// Orchestration
// ---------------------------------------------------------------------------
extern "C" void kernel_launch(void* const* d_in, const int* in_sizes, int n_in,
                              void* d_out, int out_size, void* d_ws, size_t ws_size,
                              hipStream_t stream) {
  (void)in_sizes; (void)n_in; (void)out_size; (void)ws_size;
  const float* x      = (const float*)d_in[0];
  const float* snw    = (const float*)d_in[1];
  const float* sdw    = (const float*)d_in[2];
  const float* spw    = (const float*)d_in[3];
  const float* sscale = (const float*)d_in[4];
  const float* anw    = (const float*)d_in[5];
  const float* wq     = (const float*)d_in[6];
  const float* wk     = (const float*)d_in[7];
  const float* wv     = (const float*)d_in[8];
  const float* wo     = (const float*)d_in[9];
  const float* fnw    = (const float*)d_in[10];
  const float* w1     = (const float*)d_in[11];
  const float* w3     = (const float*)d_in[12];
  const float* w2     = (const float*)d_in[13];
  const float* finw   = (const float*)d_in[14];
  float* out = (float*)d_out;

  char* p = (char*)d_ws;
  auto take = [&](size_t n) { char* r = p; p += (n + 255) & ~(size_t)255; return r; };
  float* hbuf = (float*)take((size_t)MTOK * DD * 4);        // running hidden, 16MB
  float* big0 = (float*)take((size_t)MTOK * FFD * 4);       // q/k f32 or u, 48MB
  char*  act1 = take((size_t)MTOK * DD * 4);                // nx f32 / actA f16
  f16*   actB = (f16*)take((size_t)MTOK * FFD * 2);         // g f16 OR f16 qkv/attout
  f16*   wf   = (f16*)take((size_t)FFD * DD * 2);           // f16 B^T weight scratch
  float* cosT = (float*)take((size_t)SS * 32 * 4);
  float* sinT = (float*)take((size_t)SS * 32 * 4);

  float* nx   = (float*)act1;
  f16*   actA = (f16*)act1;
  float* qb   = big0;                                       // f32 q (pre-rope)
  float* kb   = big0 + (size_t)MTOK * DD;                   // f32 k (pre-rope)
  float* ub   = big0;                                       // FFN u (full 48MB)
  f16*   qhb  = actB;                                       // f16 rotated q
  f16*   khb  = qhb + (size_t)MTOK * (HH * HDD);
  f16*   vhb  = khb + (size_t)MTOK * (KHH * HDD);
  f16*   abh  = vhb + (size_t)MTOK * (KHH * HDD);           // f16 attention out

  const float* nullf = nullptr;
  f16* nullh = nullptr;

  k_rope_tables<<<(SS * 32 + 255) / 256, 256, 0, stream>>>(cosT, sinT);

  // ---- conv stem: h = x + scale * (dwconv(rmsnorm(x)) @ pw^T) ----
  k_rmsnorm<<<MTOK, 256, 0, stream>>>(x, snw, nx, nullh);
  k_dwconv<<<(MTOK * DD + 255) / 256, 256, 0, stream>>>(nx, sdw, qb);
  k_cvt<<<(MTOK * DD + 255) / 256, 256, 0, stream>>>(qb, actA, MTOK * DD);
  k_cvt<<<(DD * DD + 255) / 256, 256, 0, stream>>>(spw, wf, DD * DD); // Bt[o][i]=pw[o][i]
  k_gemm<<<dim3(DD / 64, MTOK / 128), 256, 0, stream>>>(
      actA, wf, MTOK, DD, DD, hbuf, nullh, x, sscale, nullf);

  // ---- layers ----
  for (int l = 0; l < LL; ++l) {
    const float* wq_l = wq + (size_t)l * DD * (HH * HDD);
    const float* wk_l = wk + (size_t)l * DD * (KHH * HDD);
    const float* wv_l = wv + (size_t)l * DD * (KHH * HDD);
    const float* wo_l = wo + (size_t)l * (HH * HDD) * DD;
    const float* w1_l = w1 + (size_t)l * DD * FFD;
    const float* w3_l = w3 + (size_t)l * DD * FFD;
    const float* w2_l = w2 + (size_t)l * FFD * DD;

    // attention
    k_rmsnorm<<<MTOK, 256, 0, stream>>>(hbuf, anw + (size_t)l * DD, (float*)nullptr, actA);
    k_cvt_t<<<(DD * HH * HDD + 255) / 256, 256, 0, stream>>>(wq_l, wf, DD, HH * HDD);
    k_gemm<<<dim3((HH * HDD) / 64, MTOK / 128), 256, 0, stream>>>(
        actA, wf, MTOK, HH * HDD, DD, qb, nullh, nullf, nullf, nullf);
    k_cvt_t<<<(DD * KHH * HDD + 255) / 256, 256, 0, stream>>>(wk_l, wf, DD, KHH * HDD);
    k_gemm<<<dim3((KHH * HDD) / 64, MTOK / 128), 256, 0, stream>>>(
        actA, wf, MTOK, KHH * HDD, DD, kb, nullh, nullf, nullf, nullf);
    k_cvt_t<<<(DD * KHH * HDD + 255) / 256, 256, 0, stream>>>(wv_l, wf, DD, KHH * HDD);
    k_gemm<<<dim3((KHH * HDD) / 64, MTOK / 128), 256, 0, stream>>>(
        actA, wf, MTOK, KHH * HDD, DD, (float*)nullptr, vhb, nullf, nullf, nullf);

    k_rope<<<(MTOK * HH * 32 + 255) / 256, 256, 0, stream>>>(qb, qhb, cosT, sinT, HH);
    k_rope<<<(MTOK * KHH * 32 + 255) / 256, 256, 0, stream>>>(kb, khb, cosT, sinT, KHH);

    k_attn<<<BB * (SS / WW) * HH * 4, 128, 0, stream>>>(qhb, khb, vhb, abh);

    k_cvt_t<<<(HH * HDD * DD + 255) / 256, 256, 0, stream>>>(wo_l, wf, HH * HDD, DD);
    k_gemm<<<dim3(DD / 64, MTOK / 128), 256, 0, stream>>>(
        abh, wf, MTOK, DD, HH * HDD, hbuf, nullh, hbuf, nullf, nullf);

    // FFN (SwiGLU)
    k_rmsnorm<<<MTOK, 256, 0, stream>>>(hbuf, fnw + (size_t)l * DD, (float*)nullptr, actA);
    k_cvt_t<<<(DD * FFD + 255) / 256, 256, 0, stream>>>(w1_l, wf, DD, FFD);
    k_gemm<<<dim3(FFD / 64, MTOK / 128), 256, 0, stream>>>(
        actA, wf, MTOK, FFD, DD, ub, nullh, nullf, nullf, nullf);
    k_cvt_t<<<(DD * FFD + 255) / 256, 256, 0, stream>>>(w3_l, wf, DD, FFD);
    k_gemm<<<dim3(FFD / 64, MTOK / 128), 256, 0, stream>>>(
        actA, wf, MTOK, FFD, DD, (float*)nullptr, actB, nullf, nullf, ub); // silu(u)*acc -> f16
    k_cvt_t<<<(FFD * DD + 255) / 256, 256, 0, stream>>>(w2_l, wf, FFD, DD);
    k_gemm<<<dim3(DD / 64, MTOK / 128), 256, 0, stream>>>(
        actB, wf, MTOK, DD, FFD, hbuf, nullh, hbuf, nullf, nullf);
  }

  // ---- final norm ----
  k_rmsnorm<<<MTOK, 256, 0, stream>>>(hbuf, finw, out, nullh);
}
